// E3SchNetInteractionBlock_41884521070877
// MI455X (gfx1250) — matrix operand order repaired and checked by reading, not compiled, not run on hardware
//
#include <hip/hip_runtime.h>
#include <math.h>

// ---------------------------------------------------------------------------
// Problem constants (from the reference)
// ---------------------------------------------------------------------------
#define NN       32768
#define EE       524288
#define FF       16
#define S_IN     64
#define V_IN     32
#define NRBF     32

#define INV_SQRT64  0.125f
#define INV_SQRT32  0.17677669529663687f   // 1/sqrt(32)  (= 1/sqrt(V_IN) = 1/sqrt(2F))
#define INV_SQRT3   0.5773502691896258f
#define LN2F        0.6931471805599453f

#define N_FRAGS  24   // precomputed B fragments, 32 lanes x 16 bf16 each

// ---------------------------------------------------------------------------
// WMMA types / helpers (gfx1250 wave32, v_wmma_f32_16x16x32_bf16)
// ---------------------------------------------------------------------------
typedef __attribute__((ext_vector_type(16))) __bf16 v16bf;
typedef __attribute__((ext_vector_type(8)))  float  v8f;
typedef __attribute__((ext_vector_type(4)))  unsigned int u32x4;
typedef __attribute__((ext_vector_type(4)))  int i32x4;
typedef __attribute__((ext_vector_type(8)))  int i32x8;

__device__ __forceinline__ v8f wmma_bf16(v16bf a, v16bf b, v8f c) {
  return __builtin_amdgcn_wmma_f32_16x16x32_bf16(
      /*neg_a=*/false, a, /*neg_b=*/false, b,
      /*c_mod=*/(short)0, c, /*reuse_a=*/false, /*reuse_b=*/false);
}

// A-matrix (16x32 bf16, M x K): lane l holds row m = l&15.
// VGPR pair slot i: i<4 -> k = half*8 + 2i ; i>=4 -> k = 16 + half*8 + 2(i-4)
__device__ __forceinline__ int a_kidx(int i, int half) {
  return ((i < 4) ? 0 : 16) + half * 8 + 2 * (i & 3);
}
// B-matrix (32x16 bf16, K x N): lane l holds col n = l&15; j<4 -> k = half*8 + 2j ; j>=4 -> k = 16 + half*8 + 2(j-4)
__device__ __forceinline__ int b_kidx(int j, int half) {
  return ((j < 4) ? 0 : 16) + half * 8 + 2 * (j & 3);
}
// C/D (16x16 f32): lane l, vreg r: row m = r + 8*(l>>4), col n = l&15

__device__ __forceinline__ float ssp(float x) {
  return fmaxf(x, 0.f) + log1pf(expf(-fabsf(x))) - LN2F;   // softplus(x) - ln2
}

// Packed B-fragment load: 32 contiguous bytes per lane -> 2x global_load_b128
__device__ __forceinline__ v16bf load_frag(const __bf16* __restrict__ frags,
                                           int fid, int lane) {
  v16bf r;
  __builtin_memcpy(&r, frags + ((size_t)fid * 32 + lane) * 16, 32);
  return r;
}

// ---------------------------------------------------------------------------
// Tensor Data Mover: async-load a [rows x row_elems] f32 2-D tile into LDS.
// D# per CDNA5 ISA (count=1, type=2 "image", data_size=4B).
// ---------------------------------------------------------------------------
__device__ __forceinline__ void tdm_load_tile_f32(unsigned int lds_off,
                                                  const void* gaddr,
                                                  unsigned int rows,
                                                  unsigned int row_elems) {
  unsigned long long ga = (unsigned long long)(uintptr_t)gaddr;
  u32x4 g0;
  g0[0] = 1u;                                              // count=1 (valid), user mode
  g0[1] = lds_off;                                         // lds_addr
  g0[2] = (unsigned int)ga;                                // global_addr[31:0]
  g0[3] = (unsigned int)((ga >> 32) & 0x1FFFFFFu) | (2u << 30); // addr hi | type=2
  i32x8 g1;
  g1[0] = (int)(2u << 16);                                 // workgroup_mask=0, data_size=4B
  g1[1] = (int)((row_elems & 0xFFFFu) << 16);              // tensor_dim0[15:0]
  g1[2] = (int)((row_elems >> 16) | ((rows & 0xFFFFu) << 16)); // dim0 hi | tensor_dim1 lo
  g1[3] = (int)(((rows >> 16) & 0xFFFFu) | ((row_elems & 0xFFFFu) << 16)); // dim1 hi | tile_dim0
  g1[4] = (int)(rows & 0xFFFFu);                           // tile_dim1 ; tile_dim2 = 0
  g1[5] = (int)row_elems;                                  // tensor_dim0_stride[31:0]
  g1[6] = 0;                                               // stride hi ; dim1_stride lo
  g1[7] = 0;
  i32x4 z4 = {0, 0, 0, 0};
#if defined(__clang_major__) && (__clang_major__ >= 23)
  i32x8 z8 = {0, 0, 0, 0, 0, 0, 0, 0};
  __builtin_amdgcn_tensor_load_to_lds(g0, g1, z4, z4, z8, 0);
#else
  __builtin_amdgcn_tensor_load_to_lds(g0, g1, z4, z4, 0);
#endif
}

// ---------------------------------------------------------------------------
// Kernel 0: zero the scatter accumulators (n0 | n1, contiguous)
// ---------------------------------------------------------------------------
__global__ void k_zero(float* __restrict__ p, int count) {
  int i = blockIdx.x * blockDim.x + threadIdx.x;
  if (i < count) p[i] = 0.f;
}

// ---------------------------------------------------------------------------
// Kernel 0b: precompute all WMMA B fragments (packed bf16, 32B per lane).
// Frag ids: 0,1=W1_0e kt | 2=W1_1o | 3=mW1 | 4..7=mW2[:, :64] tiles (K<16)
//           8..11=W2_0e t | 12..19=W3_0e (u*2+kt) | 20,21=W2_1o t | 22,23=W3_1o t
// ---------------------------------------------------------------------------
__global__ void k_pack_frags(const float* __restrict__ W1_0e,
                             const float* __restrict__ W1_1o,
                             const float* __restrict__ mW1,
                             const float* __restrict__ mW2,
                             const float* __restrict__ W2_0e,
                             const float* __restrict__ W2_1o,
                             const float* __restrict__ W3_0e,
                             const float* __restrict__ W3_1o,
                             __bf16* __restrict__ frags) {
  const int lane = threadIdx.x & 31;
  const int wid  = threadIdx.x >> 5;
  const int n = lane & 15, half = lane >> 4;
  for (int f = wid; f < N_FRAGS; f += 8) {
    __bf16 v[16];
    for (int j = 0; j < 8; ++j) {
      int k = b_kidx(j, half);
      float lo = 0.f, hi = 0.f;
      if (f < 2) {                       // W1_0e (64x16), K-tile f
        int kk = f * 32 + k;
        lo = W1_0e[kk * FF + n]; hi = W1_0e[(kk + 1) * FF + n];
      } else if (f == 2) {               // W1_1o (32x16)
        lo = W1_1o[k * FF + n]; hi = W1_1o[(k + 1) * FF + n];
      } else if (f == 3) {               // mW1 (32x16)
        lo = mW1[k * FF + n]; hi = mW1[(k + 1) * FF + n];
      } else if (f < 8) {                // mW2 (16x96), live tiles t=0..3, K<16
        int t = f - 4;
        if (k < FF) { lo = mW2[k * 96 + t * 16 + n]; hi = mW2[(k + 1) * 96 + t * 16 + n]; }
      } else if (f < 12) {               // W2_0e (32x64), tile t
        int t = f - 8;
        lo = W2_0e[k * 64 + t * 16 + n]; hi = W2_0e[(k + 1) * 64 + t * 16 + n];
      } else if (f < 20) {               // W3_0e (64x64), u*2+kt
        int u = (f - 12) >> 1, kt = (f - 12) & 1;
        int kk = kt * 32 + k;
        lo = W3_0e[kk * 64 + u * 16 + n]; hi = W3_0e[(kk + 1) * 64 + u * 16 + n];
      } else if (f < 22) {               // W2_1o (32x32), tile t
        int t = f - 20;
        lo = W2_1o[k * 32 + t * 16 + n]; hi = W2_1o[(k + 1) * 32 + t * 16 + n];
      } else {                           // W3_1o (32x32), tile t
        int t = f - 22;
        lo = W3_1o[k * 32 + t * 16 + n]; hi = W3_1o[(k + 1) * 32 + t * 16 + n];
      }
      v[2 * j] = (__bf16)lo; v[2 * j + 1] = (__bf16)hi;
    }
    __bf16* dst = frags + ((size_t)f * 32 + lane) * 16;
    for (int i = 0; i < 16; ++i) dst[i] = v[i];
  }
}

// ---------------------------------------------------------------------------
// Kernel 1: node embedding (one wave = 16 nodes; 5 WMMAs)
// ---------------------------------------------------------------------------
__global__ void k_node_embed(const float* __restrict__ x,
                             const __bf16* __restrict__ frags,
                             float* __restrict__ s1,
                             float* __restrict__ v1) {
  const int lane  = threadIdx.x & 31;
  const int wave  = threadIdx.x >> 5;
  const int node0 = (blockIdx.x * 8 + wave) * 16;
  const int m = lane & 15, half = lane >> 4, n = m;
  const float* xr = x + (size_t)(node0 + m) * 160;

  // ---- s1 = s @ W1_0e / 8 ----
  v8f acc = {};
  for (int kt = 0; kt < 2; ++kt) {
    v16bf a;
    for (int i = 0; i < 8; ++i) {
      int k = kt * 32 + a_kidx(i, half);
      a[2*i]   = (__bf16)xr[k];
      a[2*i+1] = (__bf16)xr[k+1];
    }
    acc = wmma_bf16(a, load_frag(frags, kt, lane), acc);
  }
  for (int r = 0; r < 8; ++r) {
    int mr = r + 8 * half;
    s1[(size_t)(node0 + mr) * FF + n] = acc[r] * INV_SQRT64;
  }

  // ---- v1[n,f,d] = sum_c v[n,c,d] W1_1o[c,f] / sqrt(32) ----
  v16bf bv = load_frag(frags, 2, lane);
  for (int d = 0; d < 3; ++d) {
    v16bf a;
    for (int i = 0; i < 8; ++i) {
      int c = a_kidx(i, half);
      a[2*i]   = (__bf16)xr[S_IN + 3*c + d];
      a[2*i+1] = (__bf16)xr[S_IN + 3*(c+1) + d];
    }
    v8f cc = {};
    cc = wmma_bf16(a, bv, cc);
    for (int r = 0; r < 8; ++r) {
      int mr = r + 8 * half;
      v1[(size_t)(node0 + mr) * 48 + n * 3 + d] = cc[r] * INV_SQRT32;
    }
  }
}

// ---------------------------------------------------------------------------
// Kernel 2: edge MLP (TDM-staged f_ij + WMMA) + gather/scatter.
// One wave = 16 edges; f_ij tile arrives in LDS via tensor_load_to_lds.
// ---------------------------------------------------------------------------
__global__ void k_edges(const float* __restrict__ f_ij,
                        const float* __restrict__ rcut,
                        const float* __restrict__ Yr,
                        const __bf16* __restrict__ frags,
                        const float* __restrict__ mb1,
                        const float* __restrict__ mb2,
                        const int*   __restrict__ idx_i,
                        const int*   __restrict__ idx_j,
                        const float* __restrict__ s1,
                        const float* __restrict__ v1,
                        float* __restrict__ n0,
                        float* __restrict__ n1) {
  __shared__ float fstage[8][16][NRBF];  // 16 KB: TDM landing pad for f_ij tiles
  __shared__ float hlds[8][16][16];      //  8 KB: hidden activations
  __shared__ float wlds[8][16][64];      // 32 KB: edge weights
  const int lane = threadIdx.x & 31;
  const int wave = threadIdx.x >> 5;
  const int e0   = (blockIdx.x * 8 + wave) * 16;
  const int m = lane & 15, half = lane >> 4, n = m;

  // Kick the Tensor Data Mover: 16x32 f32 tile of f_ij -> LDS (async, TENSORcnt).
  tdm_load_tile_f32((unsigned int)(uintptr_t)&fstage[wave][0][0],
                    f_ij + (size_t)e0 * NRBF, 16u, (unsigned int)NRBF);
  // Stream the next block's tile toward L2 meanwhile (global_prefetch_b8).
  __builtin_prefetch(f_ij + ((size_t)e0 + 128) * NRBF, 0, 1);

  // Overlap: fetch B fragments + biases while the TDM runs.
  v16bf b1 = load_frag(frags, 3, lane);
  float bias1 = mb1[n];

  __builtin_amdgcn_s_wait_tensorcnt((short)0);

  // ---- h = ssp(f_ij @ mW1 + mb1) ----
  const float* frow = &fstage[wave][m][0];
  v16bf a;
  for (int i = 0; i < 8; ++i) {
    int k = a_kidx(i, half);
    a[2*i]   = (__bf16)frow[k];
    a[2*i+1] = (__bf16)frow[k+1];
  }
  v8f h = {};
  h = wmma_bf16(a, b1, h);
  for (int r = 0; r < 8; ++r)
    hlds[wave][r + 8*half][n] = ssp(h[r] + bias1);
  __syncthreads();

  // ---- W = (h @ mW2[:, :64] + mb2[:64]) * rcut ----  (K=16 zero-padded to 32)
  v16bf a2;
  for (int i = 0; i < 8; ++i) {
    int k = a_kidx(i, half);
    if (k < FF) { a2[2*i] = (__bf16)hlds[wave][m][k]; a2[2*i+1] = (__bf16)hlds[wave][m][k+1]; }
    else        { a2[2*i] = (__bf16)0.f;              a2[2*i+1] = (__bf16)0.f; }
  }
  for (int t = 0; t < 4; ++t) {
    v8f w = {};
    w = wmma_bf16(a2, load_frag(frags, 4 + t, lane), w);
    float bias2 = mb2[t * 16 + n];
    for (int r = 0; r < 8; ++r) {
      int mr = r + 8 * half;
      wlds[wave][mr][t * 16 + n] = (w[r] + bias2) * rcut[e0 + mr];
    }
  }
  __syncthreads();

  // ---- gather (idx_j) + scatter-add (idx_i): 16 edges x 16 features ----
  for (int it = 0; it < 8; ++it) {
    int pair = it * 32 + lane;
    int el   = pair >> 4;
    int f    = pair & 15;
    int e    = e0 + el;
    int nj   = idx_j[e];
    int ni   = idx_i[e];
    float y0  = Yr[(size_t)e * 4 + 0];
    float y1x = Yr[(size_t)e * 4 + 1];
    float y1y = Yr[(size_t)e * 4 + 2];
    float y1z = Yr[(size_t)e * 4 + 3];
    float sj  = s1[(size_t)nj * FF + f];
    const float* vp = v1 + (size_t)nj * 48 + f * 3;
    float vx = vp[0], vy = vp[1], vz = vp[2];
    float w0a = wlds[wave][el][f];
    float w0b = wlds[wave][el][16 + f];
    float w1a = wlds[wave][el][32 + f];
    float w1b = wlds[wave][el][48 + f];

    float o0a = sj * y0;
    float o0b = (vx * y1x + vy * y1y + vz * y1z) * INV_SQRT3;
    atomicAdd(&n0[(size_t)ni * 32 + f],      o0a * w0a);
    atomicAdd(&n0[(size_t)ni * 32 + 16 + f], o0b * w0b);

    float* n1p = n1 + (size_t)ni * 96;
    atomicAdd(&n1p[f * 3 + 0],        sj * y1x * w1a);
    atomicAdd(&n1p[f * 3 + 1],        sj * y1y * w1a);
    atomicAdd(&n1p[f * 3 + 2],        sj * y1z * w1a);
    atomicAdd(&n1p[(16 + f) * 3 + 0], vx * y0 * w1b);
    atomicAdd(&n1p[(16 + f) * 3 + 1], vy * y0 * w1b);
    atomicAdd(&n1p[(16 + f) * 3 + 2], vz * y0 * w1b);
  }
}

// ---------------------------------------------------------------------------
// Kernel 3: output projections (one wave = 16 nodes; C->A relayout via LDS)
// ---------------------------------------------------------------------------
__global__ void k_node_out(const float* __restrict__ n0,
                           const float* __restrict__ n1,
                           const __bf16* __restrict__ frags,
                           float* __restrict__ out) {
  __shared__ float s2lds[8][16][64];  // 32 KB
  __shared__ float v2lds[8][16][32];  // 16 KB
  const int lane  = threadIdx.x & 31;
  const int wave  = threadIdx.x >> 5;
  const int node0 = (blockIdx.x * 8 + wave) * 16;
  const int m = lane & 15, half = lane >> 4, n = m;

  // ---- s2 = ssp(n0 @ W2_0e * inv_sqrt32) ----
  const float* nr = n0 + (size_t)(node0 + m) * 32;
  v16bf a;
  for (int i = 0; i < 8; ++i) {
    int k = a_kidx(i, half);
    a[2*i] = (__bf16)nr[k]; a[2*i+1] = (__bf16)nr[k+1];
  }
  for (int t = 0; t < 4; ++t) {
    v8f c = {};
    c = wmma_bf16(a, load_frag(frags, 8 + t, lane), c);
    for (int r = 0; r < 8; ++r)
      s2lds[wave][r + 8*half][t * 16 + n] = ssp(c[r] * INV_SQRT32);
  }
  __syncthreads();

  // ---- s3 = s2 @ W3_0e * 0.125 ----
  for (int u = 0; u < 4; ++u) {
    v8f c = {};
    for (int kt = 0; kt < 2; ++kt) {
      v16bf aa;
      for (int i = 0; i < 8; ++i) {
        int k = kt * 32 + a_kidx(i, half);
        aa[2*i] = (__bf16)s2lds[wave][m][k]; aa[2*i+1] = (__bf16)s2lds[wave][m][k+1];
      }
      c = wmma_bf16(aa, load_frag(frags, 12 + u * 2 + kt, lane), c);
    }
    for (int r = 0; r < 8; ++r) {
      int mr = r + 8 * half;
      out[(size_t)(node0 + mr) * 160 + u * 16 + n] = c[r] * INV_SQRT64;
    }
  }

  // ---- vector path, per spatial dim ----
  v16bf b2v[2] = { load_frag(frags, 20, lane), load_frag(frags, 21, lane) };
  v16bf b3v[2] = { load_frag(frags, 22, lane), load_frag(frags, 23, lane) };
  for (int d = 0; d < 3; ++d) {
    __syncthreads();  // protect v2lds reuse across d iterations
    const float* n1r = n1 + (size_t)(node0 + m) * 96;
    v16bf av;
    for (int i = 0; i < 8; ++i) {
      int k = a_kidx(i, half);
      av[2*i] = (__bf16)n1r[k * 3 + d]; av[2*i+1] = (__bf16)n1r[(k+1) * 3 + d];
    }
    for (int t = 0; t < 2; ++t) {
      v8f c = {};
      c = wmma_bf16(av, b2v[t], c);
      for (int r = 0; r < 8; ++r)
        v2lds[wave][r + 8*half][t * 16 + n] = c[r] * INV_SQRT32;
    }
    __syncthreads();
    v16bf a3;
    for (int i = 0; i < 8; ++i) {
      int k = a_kidx(i, half);
      a3[2*i] = (__bf16)v2lds[wave][m][k]; a3[2*i+1] = (__bf16)v2lds[wave][m][k+1];
    }
    for (int t = 0; t < 2; ++t) {
      v8f c = {};
      c = wmma_bf16(a3, b3v[t], c);
      for (int r = 0; r < 8; ++r) {
        int mr = r + 8 * half;
        out[(size_t)(node0 + mr) * 160 + S_IN + (t * 16 + n) * 3 + d] = c[r] * INV_SQRT32;
      }
    }
  }
}

// ---------------------------------------------------------------------------
// Launch
// ---------------------------------------------------------------------------
extern "C" void kernel_launch(void* const* d_in, const int* in_sizes, int n_in,
                              void* d_out, int out_size, void* d_ws, size_t ws_size,
                              hipStream_t stream) {
  const float* x     = (const float*)d_in[0];
  const float* f_ij  = (const float*)d_in[1];
  const float* rcut  = (const float*)d_in[2];
  const float* Yr    = (const float*)d_in[3];
  const float* W1_0e = (const float*)d_in[4];
  const float* W1_1o = (const float*)d_in[5];
  const float* mW1   = (const float*)d_in[6];
  const float* mb1   = (const float*)d_in[7];
  const float* mW2   = (const float*)d_in[8];
  const float* mb2   = (const float*)d_in[9];
  const float* W2_0e = (const float*)d_in[10];
  const float* W2_1o = (const float*)d_in[11];
  const float* W3_0e = (const float*)d_in[12];
  const float* W3_1o = (const float*)d_in[13];
  const int*   idx_i = (const int*)d_in[14];
  const int*   idx_j = (const int*)d_in[15];
  float* out = (float*)d_out;

  float* ws = (float*)d_ws;
  float*  s1    = ws;                        // N*16 f32
  float*  v1    = ws + (size_t)NN * 16;      // N*48 f32
  float*  n0    = ws + (size_t)NN * 64;      // N*32 f32
  float*  n1    = ws + (size_t)NN * 96;      // N*96 f32
  __bf16* frags = (__bf16*)(ws + (size_t)NN * 192);  // 24*32*16 bf16 (24 KB)

  // zero n0|n1 (contiguous N*128 floats)
  {
    int zcount = NN * 128;
    k_zero<<<(zcount + 255) / 256, 256, 0, stream>>>(n0, zcount);
  }
  // pack all WMMA B fragments once per launch
  k_pack_frags<<<1, 256, 0, stream>>>(W1_0e, W1_1o, mW1, mW2,
                                      W2_0e, W2_1o, W3_0e, W3_1o, frags);
  // node embedding: 2048 node-tiles / 8 waves per block
  k_node_embed<<<NN / 128, 256, 0, stream>>>(x, frags, s1, v1);
  // edge MLP + scatter: 32768 edge-tiles / 8 waves per block
  k_edges<<<EE / 128, 256, 0, stream>>>(f_ij, rcut, Yr, frags, mb1, mb2,
                                        idx_i, idx_j, s1, v1, n0, n1);
  // output projections
  k_node_out<<<NN / 128, 256, 0, stream>>>(n0, n1, frags, out);
}
